// DensePool_59416577573009
// MI455X (gfx1250) — matrix operand order repaired and checked by reading, not compiled, not run on hardware
//
#include <hip/hip_runtime.h>
#include <stdint.h>
#include <stddef.h>

#define B_ 4
#define N_ 1024
#define F_ 256

typedef __attribute__((ext_vector_type(16))) _Float16 v16h;
typedef __attribute__((ext_vector_type(8)))  _Float16 v8h;
typedef __attribute__((ext_vector_type(8)))  float    v8f;
typedef __attribute__((ext_vector_type(4)))  int      v4i;

// --- gfx1250 async global->LDS path (guarded; fallback = ld/st through VGPRs) ---
#if defined(__has_builtin)
#  if __has_builtin(__builtin_amdgcn_global_load_async_to_lds_b128) && \
      __has_builtin(__builtin_amdgcn_s_wait_asynccnt)
#    define HAVE_ASYNC_LDS 1
#  endif
#endif
#ifndef HAVE_ASYNC_LDS
#  define HAVE_ASYNC_LDS 0
#endif

// ---------------------------------------------------------------------------
// WMMA operand loaders (CDNA5 16x16x32 f16 layouts, wave32)
// A (16x32, MxK): lane l -> M = l&15 ; VGPR0-3: K = kb..kb+7, VGPR4-7: K = 16+kb..+7
//                 where kb = (l>=16) ? 8 : 0.  Both halves contiguous -> b128 loads.
// B (32x16, KxN): lane l -> N = l&15 ; same K order as A (kb..kb+7, 16+kb..+7).
// ---------------------------------------------------------------------------
__device__ __forceinline__ v16h wmma_load_A(const _Float16* row, int kk, int sel) {
  v16h a;
  ((v8h*)&a)[0] = *(const v8h*)(row + kk + sel * 8);
  ((v8h*)&a)[1] = *(const v8h*)(row + kk + 16 + sel * 8);
  return a;
}
__device__ __forceinline__ v16h wmma_load_B(const _Float16* row, int kk, int sel) {
  v16h b;
  ((v8h*)&b)[0] = *(const v8h*)(row + kk + sel * 8);
  ((v8h*)&b)[1] = *(const v8h*)(row + kk + 16 + sel * 8);
  return b;
}

// Stage a contiguous 16 x N_ f16 strip (32 KB) into LDS, cooperatively by the
// 128-thread block.  Async path: each lane issues 16-byte global->LDS DMAs
// (ASYNCcnt, 16 per lane <= 63 deep), waits its own counter, then block-barrier.
__device__ __forceinline__ void stage_strip_to_lds(_Float16* lds, const _Float16* gsrc) {
  const int tid = threadIdx.x;  // 0..127
#if HAVE_ASYNC_LDS
  v4i* g = (v4i*)(gsrc);                   // 16-byte granules (int4 pointee per builtin proto)
  v4i* l = (v4i*)(lds);
  const int total = 16 * N_ * 2 / 16;      // 2048 granules
  for (int i = tid; i < total; i += 128) {
    __builtin_amdgcn_global_load_async_to_lds_b128(g + i, l + i, 0, 0);
  }
  __builtin_amdgcn_s_wait_asynccnt(0);
#else
  v8h* ld = (v8h*)lds;
  const v8h* gs = (const v8h*)gsrc;
  for (int i = tid; i < 16 * N_ / 8; i += 128) ld[i] = gs[i];
#endif
  __syncthreads();
}

// ---------------------------------------------------------------------------
// Prep: fp16 mirrors.  adjh = f16(adj);  adj1 = mask_row ? adj + I : 0
// stored both row-major (adj1h) and transposed (adj1Th).
// ---------------------------------------------------------------------------
__global__ void k_prep_adj(const float* __restrict__ adj, const uint8_t* __restrict__ node_mask,
                           _Float16* __restrict__ adjh, _Float16* __restrict__ adj1h,
                           _Float16* __restrict__ adj1Th) {
  size_t idx = (size_t)blockIdx.x * blockDim.x + threadIdx.x;
  if (idx >= (size_t)B_ * N_ * N_) return;
  int j = (int)(idx % N_);
  size_t r = idx / N_;
  int i = (int)(r % N_);
  int b = (int)(r / N_);
  float av = adj[idx];
  adjh[idx] = (_Float16)av;
  float a1 = node_mask[b * N_ + i] ? (av + (i == j ? 1.0f : 0.0f)) : 0.0f;
  adj1h[idx] = (_Float16)a1;
  adj1Th[(size_t)b * N_ * N_ + (size_t)j * N_ + i] = (_Float16)a1;
}

__global__ void k_prep_xT(const float* __restrict__ x, _Float16* __restrict__ xTh) {
  size_t idx = (size_t)blockIdx.x * blockDim.x + threadIdx.x;
  if (idx >= (size_t)B_ * N_ * F_) return;
  int f = (int)(idx % F_);
  size_t r = idx / F_;
  int n = (int)(r % N_);
  int b = (int)(r / N_);
  xTh[(size_t)b * F_ * N_ + (size_t)f * N_ + n] = (_Float16)x[idx];
}

// ---------------------------------------------------------------------------
// adj2[b] = adj1[b] @ adj[b]   (f16 WMMA, exact: operands are {0,1,2})
// 4 waves/block compute 4 tiles sharing the same 16-row A strip -> A staged in
// LDS via async copies; B read from global using adj's symmetry (row reads).
// ---------------------------------------------------------------------------
__global__ void __launch_bounds__(128) k_wmma_adj2(const _Float16* __restrict__ adj1h,
                                                   const _Float16* __restrict__ adjh,
                                                   float* __restrict__ adj2) {
  __shared__ _Float16 lA[16 * N_];          // 32 KB, shared A strip
  const int lane = threadIdx.x & 31;
  const int wave = threadIdx.x >> 5;
  const int tilesPerB = (N_ / 16) * (N_ / 16);
  const int tile0 = blockIdx.x * 4;          // aligned group: same b, same i0
  const int b  = tile0 / tilesPerB;
  const int t0 = tile0 % tilesPerB;
  const int i0 = (t0 >> 6) << 4;             // / (N_/16) * 16
  const int j0 = ((t0 & 63) + wave) << 4;

  stage_strip_to_lds(lA, adj1h + (size_t)b * N_ * N_ + (size_t)i0 * N_);

  const int m = lane & 15;
  const int sel = lane >> 4;
  const _Float16* arow = lA + (size_t)m * N_;                                  // LDS
  const _Float16* brow = adjh + (size_t)b * N_ * N_ + (size_t)(j0 + m) * N_;   // global

  v8f acc = {};
  for (int kk = 0; kk < N_; kk += 32) {
    __builtin_prefetch(brow + kk + 256, 0, 1);   // global_prefetch_b8
    v16h a  = wmma_load_A(arow, kk, sel);        // ds_load_b128 x2
    v16h bv = wmma_load_B(brow, kk, sel);        // global_load_b128 x2
    acc = __builtin_amdgcn_wmma_f32_16x16x32_f16(false, a, false, bv, (short)0, acc,
                                                 false, false);
  }
  float* dst = adj2 + (size_t)b * N_ * N_;
  const int col = j0 + m;
  const int rbase = i0 + sel * 8;
#pragma unroll
  for (int r = 0; r < 8; ++r) dst[(size_t)(rbase + r) * N_ + col] = acc[r];
}

// ---------------------------------------------------------------------------
// y[b] = adj1[b]^T @ x[b]  -> y[B,N,F] f32  (gathered later by perm)
// Same LDS-staged A strip (from adj1Th); B from xTh.
// ---------------------------------------------------------------------------
__global__ void __launch_bounds__(128) k_wmma_pool(const _Float16* __restrict__ adj1Th,
                                                   const _Float16* __restrict__ xTh,
                                                   float* __restrict__ y) {
  __shared__ _Float16 lA[16 * N_];          // 32 KB
  const int lane = threadIdx.x & 31;
  const int wave = threadIdx.x >> 5;
  const int tilesPerB = (N_ / 16) * (F_ / 16);
  const int tile0 = blockIdx.x * 4;          // aligned group: same b, same c0
  const int b  = tile0 / tilesPerB;
  const int t0 = tile0 % tilesPerB;
  const int c0 = (t0 / (F_ / 16)) * 16;
  const int f0 = ((t0 % (F_ / 16)) + wave) * 16;

  stage_strip_to_lds(lA, adj1Th + (size_t)b * N_ * N_ + (size_t)c0 * N_);

  const int m = lane & 15;
  const int sel = lane >> 4;
  const _Float16* arow = lA + (size_t)m * N_;
  const _Float16* brow = xTh + (size_t)b * F_ * N_ + (size_t)(f0 + m) * N_;

  v8f acc = {};
  for (int kk = 0; kk < N_; kk += 32) {
    __builtin_prefetch(brow + kk + 256, 0, 1);
    v16h a  = wmma_load_A(arow, kk, sel);
    v16h bv = wmma_load_B(brow, kk, sel);
    acc = __builtin_amdgcn_wmma_f32_16x16x32_f16(false, a, false, bv, (short)0, acc,
                                                 false, false);
  }
  float* dst = y + (size_t)b * N_ * F_;
  const int col = f0 + m;
  const int rbase = c0 + sel * 8;
#pragma unroll
  for (int r = 0; r < 8; ++r) dst[(size_t)(rbase + r) * F_ + col] = acc[r];
}

// ---------------------------------------------------------------------------
// column degree of adj1: colsum[b,c] = sum_n adj1[b,n,c] = rowsum of adj1T
// ---------------------------------------------------------------------------
__global__ void k_colsum(const _Float16* __restrict__ adj1Th, float* __restrict__ colsum) {
  int idx = blockIdx.x * blockDim.x + threadIdx.x;
  if (idx >= B_ * N_) return;
  const _Float16* row = adj1Th + (size_t)idx * N_;
  float s = 0.f;
  for (int n = 0; n < N_; ++n) s += (float)row[n];
  colsum[idx] = s;
}

// ---------------------------------------------------------------------------
// Greedy K-MIS selection (serial while-loop), one 1024-thread block per batch.
// excluded = adj1 != 0 (read from adj1h row), included = adj2 != 0.
// ---------------------------------------------------------------------------
__global__ void __launch_bounds__(1024) k_select(const _Float16* __restrict__ adj1h,
                                                 const float* __restrict__ adj2,
                                                 const float* __restrict__ order,
                                                 const uint8_t* __restrict__ node_mask,
                                                 uint8_t* __restrict__ selected) {
  __shared__ float rv[1024];
  __shared__ int   ri[1024];
  __shared__ int   s_idx;
  __shared__ int   s_any_av;
  __shared__ int   s_any_fr;

  const int b = blockIdx.x;
  const int tid = threadIdx.x;
  const float INF = __builtin_inff();

  const float ordv = order[b * N_ + tid];
  uint8_t av  = node_mask[b * N_ + tid] ? 1 : 0;
  uint8_t fr  = 0;
  uint8_t sel = 0;

  // idx0 = argmin(order) (first-occurrence tie break)
  rv[tid] = ordv; ri[tid] = tid;
  __syncthreads();
  for (int s = 512; s > 0; s >>= 1) {
    if (tid < s) {
      float ov = rv[tid + s]; int oi = ri[tid + s];
      if (ov < rv[tid] || (ov == rv[tid] && oi < ri[tid])) { rv[tid] = ov; ri[tid] = oi; }
    }
    __syncthreads();
  }
  if (tid == 0) s_idx = ri[0];
  __syncthreads();
  fr = (tid == s_idx) ? 1 : 0;

  const _Float16* a1b = adj1h + (size_t)b * N_ * N_;
  const float*    a2b = adj2  + (size_t)b * N_ * N_;

  for (int iter = 0; iter < 2 * N_ + 4; ++iter) {
    if (tid == 0) { s_any_av = 0; s_any_fr = 0; }
    __syncthreads();
    if (av) s_any_av = 1;          // benign same-value race
    if (fr) s_any_fr = 1;
    __syncthreads();
    if (!s_any_av) break;          // uniform exit

    const int idx = s_idx;
    const int frAnyBefore = s_any_fr;
    const uint8_t cur = (tid == idx) ? 1 : 0;
    sel |= cur;
    const uint8_t exc = ((float)a1b[(size_t)idx * N_ + tid] != 0.0f) ? 1 : 0;
    const uint8_t inc = (a2b[(size_t)idx * N_ + tid] != 0.0f) ? 1 : 0;

    av = (uint8_t)(av && !cur && !exc);
    fr = (uint8_t)((fr || inc || !frAnyBefore) && av);

    rv[tid] = fr ? ordv : INF; ri[tid] = tid;
    __syncthreads();
    for (int s = 512; s > 0; s >>= 1) {
      if (tid < s) {
        float ov = rv[tid + s]; int oi = ri[tid + s];
        if (ov < rv[tid] || (ov == rv[tid] && oi < ri[tid])) { rv[tid] = ov; ri[tid] = oi; }
      }
      __syncthreads();
    }
    if (tid == 0) { if (rv[0] < INF) s_idx = ri[0]; }  // keep idx if frontier empty
    __syncthreads();
  }
  selected[b * N_ + tid] = sel;
}

// ---------------------------------------------------------------------------
// top_k(selected as 0/1 floats) == stable partition: selected indices (asc),
// then unselected (asc).  mask_f[k] = k < count.  Also writes bool-mask output.
// ---------------------------------------------------------------------------
__global__ void __launch_bounds__(1024) k_partition(const uint8_t* __restrict__ selected,
                                                    int* __restrict__ perm,
                                                    float* __restrict__ maskf,
                                                    float* __restrict__ out_mask) {
  __shared__ int s_cnt;
  const int b = blockIdx.x;
  const int tid = threadIdx.x;
  if (tid == 0) {
    int c = 0;
    for (int n = 0; n < N_; ++n) if (selected[b * N_ + n])  perm[b * N_ + c++] = n;
    s_cnt = c;
    for (int n = 0; n < N_; ++n) if (!selected[b * N_ + n]) perm[b * N_ + c++] = n;
  }
  __syncthreads();
  float mk = (tid < s_cnt) ? 1.0f : 0.0f;
  maskf[b * N_ + tid] = mk;
  out_mask[b * N_ + tid] = mk;
}

// x_p[b,k,f] = y[b, perm[k], f] * mask_f[k]
__global__ void k_gather_x(const float* __restrict__ y, const int* __restrict__ perm,
                           const float* __restrict__ maskf, float* __restrict__ out_x) {
  size_t idx = (size_t)blockIdx.x * blockDim.x + threadIdx.x;
  if (idx >= (size_t)B_ * N_ * F_) return;
  int f = (int)(idx % F_);
  size_t r = idx / F_;
  int k = (int)(r % N_);
  int b = (int)(r / N_);
  int p = perm[b * N_ + k];
  out_x[idx] = y[(size_t)b * N_ * F_ + (size_t)p * F_ + f] * maskf[b * N_ + k];
}

// pos_p[b,k,:] = (sum_n adj1[b,n,perm[k]] * pos[b,n,:]) / deg * mask_f
__global__ void __launch_bounds__(128) k_pos(const _Float16* __restrict__ adj1Th,
                                             const float* __restrict__ pos,
                                             const int* __restrict__ perm,
                                             const float* __restrict__ maskf,
                                             const float* __restrict__ colsum,
                                             float* __restrict__ out_pos) {
  __shared__ float r0[128], r1[128], r2[128];
  const int bk = blockIdx.x;
  const int b = bk >> 10;          // / N_
  const int k = bk & (N_ - 1);
  const int tid = threadIdx.x;
  const int p = perm[b * N_ + k];
  const _Float16* arow = adj1Th + (size_t)b * N_ * N_ + (size_t)p * N_;
  const float* posb = pos + (size_t)b * N_ * 3;
  float s0 = 0.f, s1 = 0.f, s2 = 0.f;
  for (int n = tid; n < N_; n += 128) {
    float w = (float)arow[n];
    if (w != 0.0f) {
      s0 += w * posb[n * 3 + 0];
      s1 += w * posb[n * 3 + 1];
      s2 += w * posb[n * 3 + 2];
    }
  }
  r0[tid] = s0; r1[tid] = s1; r2[tid] = s2;
  __syncthreads();
  for (int s = 64; s > 0; s >>= 1) {
    if (tid < s) { r0[tid] += r0[tid + s]; r1[tid] += r1[tid + s]; r2[tid] += r2[tid + s]; }
    __syncthreads();
  }
  if (tid == 0) {
    float d = colsum[b * N_ + p];
    if (d == 0.f) d = 1.f;                 // padded cols always have deg>=1 w/ full mask
    float mk = maskf[b * N_ + k];
    float* o = out_pos + (size_t)b * N_ * 3 + (size_t)k * 3;
    o[0] = r0[0] / d * mk;
    o[1] = r1[0] / d * mk;
    o[2] = r2[0] / d * mk;
  }
}

// a[b,i,j] = adj2[b, perm[i], perm[j]] * mask_f[i] * mask_f[j]
__global__ void k_gather_a(const float* __restrict__ adj2, const int* __restrict__ perm,
                           const float* __restrict__ maskf, float* __restrict__ out_a) {
  size_t idx = (size_t)blockIdx.x * blockDim.x + threadIdx.x;
  if (idx >= (size_t)B_ * N_ * N_) return;
  int j = (int)(idx % N_);
  size_t r = idx / N_;
  int i = (int)(r % N_);
  int b = (int)(r / N_);
  int pi = perm[b * N_ + i];
  int pj = perm[b * N_ + j];
  out_a[idx] = adj2[(size_t)b * N_ * N_ + (size_t)pi * N_ + pj] *
               maskf[b * N_ + i] * maskf[b * N_ + j];
}

// ---------------------------------------------------------------------------
extern "C" void kernel_launch(void* const* d_in, const int* in_sizes, int n_in,
                              void* d_out, int out_size, void* d_ws, size_t ws_size,
                              hipStream_t stream) {
  (void)in_sizes; (void)n_in; (void)out_size; (void)ws_size;
  const float*   x     = (const float*)d_in[0];     // [B,N,F]
  const float*   adj   = (const float*)d_in[1];     // [B,N,N]
  const float*   pos   = (const float*)d_in[2];     // [B,N,3]
  const float*   order = (const float*)d_in[3];     // [B,N]
  const uint8_t* nmask = (const uint8_t*)d_in[4];   // [B,N] bool

  // output layout: x_p | pos_p | a | mask   (flat, f32)
  float* out_x    = (float*)d_out;                                   // B*N*F
  float* out_pos  = out_x + (size_t)B_ * N_ * F_;                    // B*N*3
  float* out_a    = out_pos + (size_t)B_ * N_ * 3;                   // B*N*N
  float* out_mask = out_a + (size_t)B_ * N_ * N_;                    // B*N

  // workspace carve-up (256B aligned) — total ~48 MB
  char* ws = (char*)d_ws;
  auto take = [&](size_t bytes) -> char* {
    char* p = ws;
    ws += (bytes + 255) & ~(size_t)255;
    return p;
  };
  _Float16* adjh   = (_Float16*)take((size_t)B_ * N_ * N_ * 2);
  _Float16* adj1h  = (_Float16*)take((size_t)B_ * N_ * N_ * 2);
  _Float16* adj1Th = (_Float16*)take((size_t)B_ * N_ * N_ * 2);
  _Float16* xTh    = (_Float16*)take((size_t)B_ * F_ * N_ * 2);
  float*    adj2   = (float*)take((size_t)B_ * N_ * N_ * 4);
  float*    y      = (float*)take((size_t)B_ * N_ * F_ * 4);
  float*    colsum = (float*)take((size_t)B_ * N_ * 4);
  float*    maskf  = (float*)take((size_t)B_ * N_ * 4);
  int*      perm   = (int*)take((size_t)B_ * N_ * 4);
  uint8_t*  selv   = (uint8_t*)take((size_t)B_ * N_);

  const size_t nAdj = (size_t)B_ * N_ * N_;
  const size_t nX   = (size_t)B_ * N_ * F_;

  k_prep_adj<<<(unsigned)((nAdj + 255) / 256), 256, 0, stream>>>(adj, nmask, adjh, adj1h, adj1Th);
  k_prep_xT <<<(unsigned)((nX + 255) / 256), 256, 0, stream>>>(x, xTh);

  // adj2 = adj1 @ adj : 16384 tiles, 4 waves/block sharing LDS-staged A strip
  k_wmma_adj2<<<B_ * (N_ / 16) * (N_ / 16) / 4, 128, 0, stream>>>(adj1h, adjh, adj2);
  // y = adj1^T @ x : 4096 tiles
  k_wmma_pool<<<B_ * (N_ / 16) * (F_ / 16) / 4, 128, 0, stream>>>(adj1Th, xTh, y);

  k_colsum<<<(B_ * N_ + 255) / 256, 256, 0, stream>>>(adj1Th, colsum);

  k_select<<<B_, 1024, 0, stream>>>(adj1h, adj2, order, nmask, selv);
  k_partition<<<B_, 1024, 0, stream>>>(selv, perm, maskf, out_mask);

  k_gather_x<<<(unsigned)((nX + 255) / 256), 256, 0, stream>>>(y, perm, maskf, out_x);
  k_pos<<<B_ * N_, 128, 0, stream>>>(adj1Th, pos, perm, maskf, colsum, out_pos);
  k_gather_a<<<(unsigned)((nAdj + 255) / 256), 256, 0, stream>>>(adj2, perm, maskf, out_a);
}
